// MessagePassingLayer_35708358099145
// MI455X (gfx1250) — matrix operand order repaired and checked by reading, not compiled, not run on hardware
//
#include <hip/hip_runtime.h>

#define Bsz   64
#define Nn    5000
#define Dd    128
#define TILES 79          // ceil(5000/64)
#define EPSC  1e-5f

typedef __attribute__((ext_vector_type(16))) _Float16 v16h;
typedef __attribute__((ext_vector_type(8)))  _Float16 v8h;
typedef __attribute__((ext_vector_type(8)))  float    v8f;

__device__ __forceinline__ v8f wmma16(v16h a, v16h b, v8f c) {
  // D = A(16x32 f16) * B(32x16 f16) + C(16x16 f32)
  return __builtin_amdgcn_wmma_f32_16x16x32_f16(false, a, false, b, (short)0, c, false, false);
}

// A fragment (16x32 f16, K-step base kb) from a global fp32 row.
// Lane L: row = L&15 (caller supplies that row's pointer), h = L>>4.
// element e<8 : k = kb + 8h + e ;  e>=8 : k = kb + 16 + 8h + (e-8)
__device__ __forceinline__ v16h a_frag_g(const float* __restrict__ rowp, int kb, int h) {
  const float4* p = (const float4*)(rowp + kb + 8 * h);
  float4 x0 = p[0], x1 = p[1];     // k = kb+8h .. +7
  float4 y0 = p[4], y1 = p[5];     // k = kb+8h+16 .. +23
  v16h a;
  a[0]=(_Float16)x0.x;  a[1]=(_Float16)x0.y;  a[2]=(_Float16)x0.z;  a[3]=(_Float16)x0.w;
  a[4]=(_Float16)x1.x;  a[5]=(_Float16)x1.y;  a[6]=(_Float16)x1.z;  a[7]=(_Float16)x1.w;
  a[8]=(_Float16)y0.x;  a[9]=(_Float16)y0.y;  a[10]=(_Float16)y0.z; a[11]=(_Float16)y0.w;
  a[12]=(_Float16)y1.x; a[13]=(_Float16)y1.y; a[14]=(_Float16)y1.z; a[15]=(_Float16)y1.w;
  return a;
}

// Same A fragment, from an LDS f16 row (row stride 128 halves).
__device__ __forceinline__ v16h a_frag_s(const _Float16* rowp, int kb, int h) {
  v8h lo = *(const v8h*)(rowp + kb + 8 * h);
  v8h hi = *(const v8h*)(rowp + kb + 8 * h + 16);
  v16h a;
#pragma unroll
  for (int e = 0; e < 8; ++e) { a[e] = lo[e]; a[8 + e] = hi[e]; }
  return a;
}

// B fragment (32x16 f16): lane col n = L&15, e<8: k = kb + 8h + e ; e>=8: k = kb + 16 + 8h + (e-8)
__device__ __forceinline__ v16h b_frag(const _Float16* __restrict__ w, int ldk, int ncol, int kb, int h) {
  const _Float16* p = w + (size_t)ncol * ldk + kb + 8 * h;
  v8h lo = *(const v8h*)(p);
  v8h hi = *(const v8h*)(p + 16);
  v16h r;
#pragma unroll
  for (int e = 0; e < 8; ++e) { r[e] = lo[e]; r[8 + e] = hi[e]; }
  return r;
}

// ---------------- weight fp32 -> f16 conversion ----------------
__global__ void cvt_weights(const float* __restrict__ wl, const float* __restrict__ wr,
                            const float* __restrict__ wc, const float* __restrict__ wf,
                            _Float16* __restrict__ w16) {
  int i = blockIdx.x * blockDim.x + threadIdx.x;
  if (i < 16384)       w16[i] = (_Float16)wl[i];
  else if (i < 32768)  w16[i] = (_Float16)wr[i - 16384];
  else if (i < 65536)  w16[i] = (_Float16)wc[i - 32768];
  else if (i < 81920)  w16[i] = (_Float16)wf[i - 65536];
}

// ---------------- fused message-passing main kernel ----------------
__global__ __launch_bounds__(256)
void mp_main(const int* __restrict__ nbr, const float* __restrict__ emb,
             const _Float16* __restrict__ w16,
             const float* __restrict__ b_comb, const float* __restrict__ b_ff,
             float* __restrict__ out, float* __restrict__ psum, float* __restrict__ psum2) {
  const int tile = blockIdx.x;           // 0..78
  const int b    = blockIdx.y;           // 0..63
  const int tid  = threadIdx.x;
  const int wave = tid >> 5;
  const int lane = tid & 31;
  const int mt   = wave & 3;             // m-tile (16 rows each) within 64-row block tile
  const int nh   = wave >> 2;            // which 64-feature half
  const int h    = lane >> 4;
  const int mr   = lane & 15;

  __shared__ _Float16 s_info[64 * 128];
  __shared__ _Float16 s_h[64 * 128];
  __shared__ float    s_ps[4][128];
  __shared__ float    s_ps2[4][128];

  const _Float16* wl = w16;
  const _Float16* wr = w16 + 16384;
  const _Float16* wc = w16 + 32768;      // (128, 256) row-major
  const _Float16* wf = w16 + 65536;

  // Row owned by this lane for A operands
  const int mloc   = mt * 16 + mr;                 // 0..63 in tile
  const int gc_raw = tile * 64 + mloc;             // global customer
  const int gc     = gc_raw < Nn ? gc_raw : (Nn - 1);
  const int iL = nbr[((size_t)b * Nn + gc) * 2 + 0];
  const int iR = nbr[((size_t)b * Nn + gc) * 2 + 1];
  const float* rowL = emb + ((size_t)(b * (Nn + 1) + iL)) * Dd;
  const float* rowR = emb + ((size_t)(b * (Nn + 1) + iR)) * Dd;
  const float* rowC = emb + ((size_t)(b * (Nn + 1) + 1 + gc)) * Dd;

  v8f acc[4];

  // ---- stage 1: info = left @ wl^T + right @ wr^T ----
#pragma unroll
  for (int nt = 0; nt < 4; ++nt)
#pragma unroll
    for (int r = 0; r < 8; ++r) acc[nt][r] = 0.0f;

#pragma unroll
  for (int kk = 0; kk < 4; ++kk) {
    const int kb = kk * 32;
    v16h aL = a_frag_g(rowL, kb, h);
    v16h aR = a_frag_g(rowR, kb, h);
#pragma unroll
    for (int nt = 0; nt < 4; ++nt) {
      const int nc = nh * 64 + nt * 16 + mr;
      acc[nt] = wmma16(aL, b_frag(wl, 128, nc, kb, h), acc[nt]);
      acc[nt] = wmma16(aR, b_frag(wr, 128, nc, kb, h), acc[nt]);
    }
  }
#pragma unroll
  for (int nt = 0; nt < 4; ++nt) {
    const int nc = nh * 64 + nt * 16 + mr;
#pragma unroll
    for (int r = 0; r < 8; ++r)
      s_info[(mt * 16 + 8 * h + r) * 128 + nc] = (_Float16)acc[nt][r];
  }
  __syncthreads();

  // ---- stage 2: h = relu(cust @ Wc1^T + info @ Wc2^T + b_comb) ----
#pragma unroll
  for (int nt = 0; nt < 4; ++nt) {
    const float bv = b_comb[nh * 64 + nt * 16 + mr];
#pragma unroll
    for (int r = 0; r < 8; ++r) acc[nt][r] = bv;
  }
  const _Float16* infoRow = s_info + mloc * 128;
#pragma unroll
  for (int kk = 0; kk < 4; ++kk) {
    const int kb = kk * 32;
    v16h aC = a_frag_g(rowC, kb, h);
    v16h aI = a_frag_s(infoRow, kb, h);
#pragma unroll
    for (int nt = 0; nt < 4; ++nt) {
      const int nc = nh * 64 + nt * 16 + mr;
      acc[nt] = wmma16(aC, b_frag(wc,       256, nc, kb, h), acc[nt]);  // Wc1 = w_comb[:, :128]
      acc[nt] = wmma16(aI, b_frag(wc + 128, 256, nc, kb, h), acc[nt]);  // Wc2 = w_comb[:, 128:]
    }
  }
#pragma unroll
  for (int nt = 0; nt < 4; ++nt) {
    const int nc = nh * 64 + nt * 16 + mr;
#pragma unroll
    for (int r = 0; r < 8; ++r)
      s_h[(mt * 16 + 8 * h + r) * 128 + nc] = (_Float16)fmaxf(acc[nt][r], 0.0f);
  }
  __syncthreads();

  // ---- stage 3: h2 = h @ wf^T + b_ff ; added = cust + h2 ----
#pragma unroll
  for (int nt = 0; nt < 4; ++nt) {
    const float bv = b_ff[nh * 64 + nt * 16 + mr];
#pragma unroll
    for (int r = 0; r < 8; ++r) acc[nt][r] = bv;
  }
  const _Float16* hRow = s_h + mloc * 128;
#pragma unroll
  for (int kk = 0; kk < 4; ++kk) {
    const int kb = kk * 32;
    v16h aH = a_frag_s(hRow, kb, h);
#pragma unroll
    for (int nt = 0; nt < 4; ++nt) {
      const int nc = nh * 64 + nt * 16 + mr;
      acc[nt] = wmma16(aH, b_frag(wf, 128, nc, kb, h), acc[nt]);
    }
  }

  // epilogue: residual add, store to d_out rows 1..N, deterministic column partial sums
#pragma unroll
  for (int nt = 0; nt < 4; ++nt) {
    const int nc = nh * 64 + nt * 16 + mr;
    float s = 0.0f, s2 = 0.0f;
#pragma unroll
    for (int r = 0; r < 8; ++r) {
      const int m  = mt * 16 + 8 * h + r;       // C-layout row
      const int gr = tile * 64 + m;
      if (gr < Nn) {
        const size_t ei = ((size_t)(b * (Nn + 1) + 1 + gr)) * Dd + nc;
        const float av = emb[ei] + acc[nt][r];
        out[ei] = av;
        s += av; s2 += av * av;
      }
    }
    s  += __shfl_xor(s, 16, 32);
    s2 += __shfl_xor(s2, 16, 32);
    if (h == 0) { s_ps[mt][nc] = s; s_ps2[mt][nc] = s2; }
  }
  __syncthreads();
  if (tid < 128) {
    const float S  = s_ps[0][tid]  + s_ps[1][tid]  + s_ps[2][tid]  + s_ps[3][tid];
    const float S2 = s_ps2[0][tid] + s_ps2[1][tid] + s_ps2[2][tid] + s_ps2[3][tid];
    psum [((size_t)b * TILES + tile) * Dd + tid] = S;
    psum2[((size_t)b * TILES + tile) * Dd + tid] = S2;
  }
}

// ---------------- stats reduction: mean / rstd per (b, d) ----------------
__global__ void reduce_stats(const float* __restrict__ psum, const float* __restrict__ psum2,
                             float* __restrict__ meanb, float* __restrict__ rstdb) {
  int i = blockIdx.x * blockDim.x + threadIdx.x;   // b*128 + d
  if (i >= Bsz * Dd) return;
  const int b = i >> 7, d = i & 127;
  float s = 0.0f, s2 = 0.0f;
  for (int t = 0; t < TILES; ++t) {
    s  += psum [((size_t)b * TILES + t) * Dd + d];
    s2 += psum2[((size_t)b * TILES + t) * Dd + d];
  }
  const float m = s / (float)Nn;
  const float v = s2 / (float)Nn - m * m;
  meanb[i] = m;
  rstdb[i] = rsqrtf(v + EPSC);
}

// ---------------- in-place normalize + depot passthrough ----------------
__global__ void normalize_k(float* __restrict__ out, const float* __restrict__ emb,
                            const float* __restrict__ meanb, const float* __restrict__ rstdb,
                            const float* __restrict__ nw, const float* __restrict__ nb) {
  const size_t i = (size_t)blockIdx.x * blockDim.x + threadIdx.x;
  const size_t total = (size_t)Bsz * (Nn + 1) * Dd;
  if (i >= total) return;
  const int d = (int)(i % Dd);
  const size_t rowi = i / Dd;
  const int row = (int)(rowi % (Nn + 1));
  const int b   = (int)(rowi / (Nn + 1));
  if (row == 0) {
    out[i] = emb[i];                        // depot row passthrough
  } else {
    const float x = out[i];
    out[i] = (x - meanb[b * Dd + d]) * rstdb[b * Dd + d] * nw[d] + nb[d];
  }
}

extern "C" void kernel_launch(void* const* d_in, const int* in_sizes, int n_in,
                              void* d_out, int out_size, void* d_ws, size_t ws_size,
                              hipStream_t stream) {
  // setup_inputs order: batch_size, num_customers, solution_neighbours, embeddings,
  //                     w_left, w_right, w_comb, b_comb, w_ff, b_ff, norm_w, norm_b
  const int*   nbr = (const int*)  d_in[2];
  const float* emb = (const float*)d_in[3];
  const float* wl  = (const float*)d_in[4];
  const float* wr  = (const float*)d_in[5];
  const float* wc  = (const float*)d_in[6];
  const float* bc  = (const float*)d_in[7];
  const float* wf  = (const float*)d_in[8];
  const float* bf  = (const float*)d_in[9];
  const float* nw  = (const float*)d_in[10];
  const float* nb  = (const float*)d_in[11];
  float* out = (float*)d_out;

  char* ws = (char*)d_ws;
  _Float16* w16 = (_Float16*)ws;
  size_t off = (size_t)81920 * 2;                          // f16 weights: 160 KB
  float* psum  = (float*)(ws + off); off += (size_t)Bsz * TILES * Dd * 4;
  float* psum2 = (float*)(ws + off); off += (size_t)Bsz * TILES * Dd * 4;
  float* meanb = (float*)(ws + off); off += (size_t)Bsz * Dd * 4;
  float* rstdb = (float*)(ws + off);

  cvt_weights<<<(81920 + 255) / 256, 256, 0, stream>>>(wl, wr, wc, wf, w16);

  dim3 grid(TILES, Bsz);
  mp_main<<<grid, 256, 0, stream>>>(nbr, emb, w16, bc, bf, out, psum, psum2);

  reduce_stats<<<(Bsz * Dd + 255) / 256, 256, 0, stream>>>(psum, psum2, meanb, rstdb);

  const size_t total = (size_t)Bsz * (Nn + 1) * Dd;
  normalize_k<<<(unsigned)((total + 255) / 256), 256, 0, stream>>>(out, emb, meanb, rstdb, nw, nb);
}
